// DB_Transform_2894807958202
// MI455X (gfx1250) — compile-verified, matrix operand order checked by reading, not run on hardware
//
#include <hip/hip_runtime.h>
#include <math.h>

typedef __attribute__((ext_vector_type(16))) _Float16 v16h;
typedef __attribute__((ext_vector_type(8)))  float    v8f;

#define DEV static __device__ __forceinline__

namespace {

constexpr int BB = 4, NC = 128, TT = 512, KBANDS = 30, KB = 120;
constexpr int MTOK = BB * TT;        // 2048
constexpr int RG = MTOK * KBANDS;    // 61440 rows for graph GEMMs

constexpr size_t ALG(size_t x) { return (x + 255) & ~(size_t)255; }

// ---- workspace layout (bytes) ----
constexpr size_t o_wqT   = 0;                               // [30][128][128] f16 (WqT[m][n])
constexpr size_t o_wkT   = ALG(o_wqT   + 983040);
constexpr size_t o_wvT   = ALG(o_wkT   + 983040);           // [30][64][128]
constexpr size_t o_woT   = ALG(o_wvT   + 491520);           // [30][128][64]
constexpr size_t o_gwT   = ALG(o_woT   + 491520);           // 4x [128][128] (q,k,v,o transposed)
constexpr size_t o_wc    = ALG(o_gwT   + 131072);           // conv weights f16, [n][j][c] per scale
constexpr size_t o_amask = ALG(o_wc    + 2031616);          // [30][30] f32
constexpr size_t o_lam   = ALG(o_amask + 3600);             // 1 f32
constexpr size_t o_rstat = ALG(o_lam   + 256);              // [2][512][4] f32
constexpr size_t o_xh    = ALG(o_rstat + 16384);            // [120][512][128] f16
constexpr size_t o_qh    = ALG(o_xh    + 15728640);         // [120][512][128] f16
constexpr size_t o_kh    = ALG(o_qh    + 15728640);         // [120][512][128] f16
constexpr size_t o_vT    = ALG(o_kh    + 15728640);         // [120][64][512] f16
constexpr size_t o_S1    = ALG(o_vT    + 7864320);          // [512][512] f32
constexpr size_t o_S2    = ALG(o_S1    + 1048576);
constexpr size_t o_P1    = ALG(o_S2    + 1048576);          // [512][512] f16
constexpr size_t o_P2    = ALG(o_P1    + 524288);
constexpr size_t o_obh   = ALG(o_P2    + 524288);           // [512][64] f16
constexpr size_t o_h1h   = ALG(o_obh   + 65536);            // [4][30][128][512] f16
constexpr size_t o_xgh   = ALG(o_h1h   + 15728640);         // [2048][30][128] f16
// overlays (lifetimes disjoint):
constexpr size_t o_gq    = o_xh;                            // [61440][128] f16
constexpr size_t o_gk    = o_qh;
constexpr size_t o_gv    = o_kh;
constexpr size_t o_gS    = o_h1h;                           // [2048][4][30][30] f32 (29.5MB over 31.4MB)
constexpr size_t o_ogh   = ALG(o_xgh   + 15728640);         // [61440][128] f16

// ---- WMMA helpers ----
DEV v8f wmma16(v16h a, v16h b, v8f c) {
  return __builtin_amdgcn_wmma_f32_16x16x32_f16(false, a, false, b, (short)0, c, false, false);
}

// Load a 16x32 f16 fragment (A-style; B loaded from B^T with same pattern).
// ISA 7.12.2: lanes 0-15 hold row (lane&15), K {0..7,16..23}; lanes 16-31 K {8..15,24..31}.
DEV v16h ld_frag(const _Float16* __restrict__ p, int ld, int lane) {
  const _Float16* row = p + (size_t)(lane & 15) * (size_t)ld + (size_t)((lane >> 4) << 3);
  v16h r;
#pragma unroll
  for (int i = 0; i < 8; ++i) r[i] = row[i];
#pragma unroll
  for (int i = 0; i < 8; ++i) r[8 + i] = row[16 + i];
  return r;
}

DEV float bred(float v, volatile float* red, int tid, int op) {
  red[tid] = v; __syncthreads();
  for (int s = 128; s > 0; s >>= 1) {
    if (tid < s) {
      float a = red[tid], b = red[tid + s];
      red[tid] = (op == 0) ? (a + b) : (op == 1 ? fmaxf(a, b) : fminf(a, b));
    }
    __syncthreads();
  }
  float r = red[0]; __syncthreads(); return r;
}

} // namespace

// ================= weight / input packing =================

// dst[b][c][r] = src[b][r][c]  (transpose last two dims, f32->f16)
__global__ void dbt_tcvt(const float* __restrict__ src, _Float16* __restrict__ dst,
                         int Bn, int R, int C) {
  int idx = blockIdx.x * 256 + threadIdx.x;
  int total = Bn * R * C;
  if (idx >= total) return;
  int c = idx % C;
  int r = (idx / C) % R;
  int b = idx / (C * R);
  dst[(size_t)b * R * C + (size_t)c * R + r] = (_Float16)src[idx];
}

// xh[kb][t][n] = x[b][n][t][k], kb = k*4 + b
__global__ void dbt_pack_x(const float* __restrict__ x, _Float16* __restrict__ xh) {
  int idx = blockIdx.x * 256 + threadIdx.x;
  if (idx >= KB * TT * NC) return;
  int n = idx & 127;
  int t = (idx >> 7) & 511;
  int kb = idx >> 16;
  int b = kb & 3, k = kb >> 2;
  xh[idx] = (_Float16)x[(((size_t)b * NC + n) * TT + t) * KBANDS + k];
}

// ================= stage 1: differential band attention =================

__global__ __launch_bounds__(32) void dbt_qkv(const _Float16* __restrict__ xh,
    const _Float16* __restrict__ wqT, const _Float16* __restrict__ wkT,
    const _Float16* __restrict__ wvT,
    _Float16* __restrict__ qh, _Float16* __restrict__ kh, _Float16* __restrict__ vT) {
  const int lane = threadIdx.x & 31;
  const int tt = blockIdx.x, oi = blockIdx.y, kb = blockIdx.z;
  const int band = kb >> 2;
  const _Float16* A0 = xh + (size_t)kb * TT * NC + (size_t)tt * 16 * NC;
  const _Float16* BT;
  if (oi < 8)       BT = wqT + (size_t)band * NC * NC + (size_t)oi * 16 * NC;
  else if (oi < 16) BT = wkT + (size_t)band * NC * NC + (size_t)(oi - 8) * 16 * NC;
  else              BT = wvT + (size_t)band * 64 * NC + (size_t)(oi - 16) * 16 * NC;
  v8f acc = {};
#pragma unroll
  for (int ks = 0; ks < 4; ++ks)
    acc = wmma16(ld_frag(A0 + ks * 32, NC, lane), ld_frag(BT + ks * 32, NC, lane), acc);
  const int colL = lane & 15;
  const int rb = tt * 16 + ((lane >> 4) << 3);
  if (oi < 16) {
    _Float16* out = (oi < 8 ? qh : kh) + (size_t)kb * TT * NC;
    int col = ((oi & 7) << 4) + colL;
#pragma unroll
    for (int i = 0; i < 8; ++i) out[(size_t)(rb + i) * NC + col] = (_Float16)acc[i];
  } else {
    _Float16* out = vT + (size_t)kb * 64 * TT;
    int col = ((oi - 16) << 4) + colL;
#pragma unroll
    for (int i = 0; i < 8; ++i) out[(size_t)col * TT + (rb + i)] = (_Float16)acc[i];
  }
}

__global__ __launch_bounds__(32) void dbt_scores(const _Float16* __restrict__ qh,
    const _Float16* __restrict__ kh, float* __restrict__ S1, float* __restrict__ S2, int kb) {
  const int lane = threadIdx.x & 31;
  const int tt = blockIdx.x, st = blockIdx.y, mh = blockIdx.z; // mh: 0=first half, 1=second
  const _Float16* A0 = qh + (size_t)kb * TT * NC + (size_t)tt * 16 * NC + mh * 64;
  const _Float16* BT = kh + (size_t)kb * TT * NC + (size_t)st * 16 * NC + mh * 64;
  v8f acc = {};
#pragma unroll
  for (int ks = 0; ks < 2; ++ks)
    acc = wmma16(ld_frag(A0 + ks * 32, NC, lane), ld_frag(BT + ks * 32, NC, lane), acc);
  float* S = mh ? S2 : S1;
  const int col = st * 16 + (lane & 15);
  const int rb = tt * 16 + ((lane >> 4) << 3);
#pragma unroll
  for (int i = 0; i < 8; ++i) S[(size_t)(rb + i) * TT + col] = acc[i] * 0.25f; // 1/sqrt(16)
}

__global__ __launch_bounds__(256) void dbt_softmax(const float* __restrict__ S1,
    const float* __restrict__ S2, _Float16* __restrict__ P1, _Float16* __restrict__ P2,
    float* __restrict__ rstat) {
  __shared__ float red[256];
  const int row = blockIdx.x, m = blockIdx.y, tid = threadIdx.x;
  const float* S = (m ? S2 : S1) + (size_t)row * TT;
  _Float16* P = (m ? P2 : P1) + (size_t)row * TT;
  float v0 = S[tid], v1 = S[tid + 256];
  float mx = bred(fmaxf(v0, v1), red, tid, 1);
  float e0 = __expf(v0 - mx), e1 = __expf(v1 - mx);
  float sum = bred(e0 + e1, red, tid, 0);
  float inv = 1.f / sum;
  float p0 = e0 * inv, p1 = e1 * inv;
  P[tid] = (_Float16)p0; P[tid + 256] = (_Float16)p1;
  float sp  = bred(p0 + p1, red, tid, 0);
  float ss  = bred(p0 * p0 + p1 * p1, red, tid, 0);
  float pmx = bred(fmaxf(p0, p1), red, tid, 1);
  float pmn = bred(fminf(p0, p1), red, tid, 2);
  if (tid == 0) {
    float* r = rstat + ((size_t)m * TT + row) * 4;
    r[0] = sp; r[1] = ss; r[2] = pmx; r[3] = pmn;
  }
}

__global__ __launch_bounds__(256) void dbt_lambda(const float* __restrict__ rstat,
    const float* __restrict__ lw1, const float* __restrict__ lb1,
    const float* __restrict__ lw2, const float* __restrict__ lb2,
    float* __restrict__ lam, int kb) {
  __shared__ float red[256];
  __shared__ float st[8];
  const int tid = threadIdx.x;
  const int band = kb >> 2;
  for (int m = 0; m < 2; ++m) {
    float s = 0.f, q = 0.f, mx = -1e30f, mn = 1e30f;
    for (int r = tid; r < TT; r += 256) {
      const float* rr = rstat + ((size_t)m * TT + r) * 4;
      s += rr[0]; q += rr[1]; mx = fmaxf(mx, rr[2]); mn = fminf(mn, rr[3]);
    }
    float S = bred(s, red, tid, 0);
    float Q = bred(q, red, tid, 0);
    float MX = bred(mx, red, tid, 1);
    float MN = bred(mn, red, tid, 2);
    if (tid == 0) {
      const float n = (float)TT * (float)TT;
      float mean = S / n;
      float var = (Q - n * mean * mean) / (n - 1.f);
      st[m * 4 + 0] = mean;
      st[m * 4 + 1] = sqrtf(fmaxf(var, 0.f));
      st[m * 4 + 2] = MX;
      st[m * 4 + 3] = MN;
    }
    __syncthreads();
  }
  if (tid == 0) {
    float z = lb2[band];
    for (int o = 0; o < 16; ++o) {
      float h = lb1[band * 16 + o];
      for (int i = 0; i < 8; ++i) h += st[i] * lw1[band * 128 + i * 16 + o];
      h = fmaxf(h, 0.f);
      z += h * lw2[band * 16 + o];
    }
    lam[0] = 1.f / (1.f + __expf(-z));
  }
}

__global__ __launch_bounds__(32) void dbt_pv(const _Float16* __restrict__ P1,
    const _Float16* __restrict__ P2, const _Float16* __restrict__ vT,
    const float* __restrict__ lam, _Float16* __restrict__ obh, int kb) {
  const int lane = threadIdx.x & 31;
  const int tt = blockIdx.x, dt = blockIdx.y;
  const _Float16* A1 = P1 + (size_t)tt * 16 * TT;
  const _Float16* A2 = P2 + (size_t)tt * 16 * TT;
  const _Float16* BT = vT + (size_t)kb * 64 * TT + (size_t)dt * 16 * TT;
  v8f acc1 = {}, acc2 = {};
  for (int ks = 0; ks < 16; ++ks) {
    v16h b = ld_frag(BT + ks * 32, TT, lane);
    acc1 = wmma16(ld_frag(A1 + ks * 32, TT, lane), b, acc1);
    acc2 = wmma16(ld_frag(A2 + ks * 32, TT, lane), b, acc2);
  }
  const float L = lam[0];
  const int col = dt * 16 + (lane & 15);
  const int rb = tt * 16 + ((lane >> 4) << 3);
#pragma unroll
  for (int i = 0; i < 8; ++i)
    obh[(size_t)(rb + i) * 64 + col] = (_Float16)(acc1[i] - L * acc2[i]);
}

__global__ __launch_bounds__(32) void dbt_wo(const _Float16* __restrict__ obh,
    const _Float16* __restrict__ woT, const float* __restrict__ bo,
    _Float16* __restrict__ h1h, int kb) {
  const int lane = threadIdx.x & 31;
  const int tt = blockIdx.x, nt = blockIdx.y;
  const int band = kb >> 2, b = kb & 3;
  const _Float16* A0 = obh + (size_t)tt * 16 * 64;
  const _Float16* BT = woT + (size_t)band * NC * 64 + (size_t)nt * 16 * 64;
  v8f acc = {};
#pragma unroll
  for (int ks = 0; ks < 2; ++ks)
    acc = wmma16(ld_frag(A0 + ks * 32, 64, lane), ld_frag(BT + ks * 32, 64, lane), acc);
  const int col = nt * 16 + (lane & 15);
  const int rb = tt * 16 + ((lane >> 4) << 3);
  const float bias = bo[band * NC + col];
#pragma unroll
  for (int i = 0; i < 8; ++i)
    h1h[(((size_t)b * KBANDS + band) * NC + col) * TT + (rb + i)] = (_Float16)(acc[i] + bias);
}

// ================= stage 2: multiscale conv fusion (implicit GEMM) =================

__global__ __launch_bounds__(32) void dbt_conv(const _Float16* __restrict__ h1h,
    const _Float16* __restrict__ wc,
    const float* __restrict__ cb2, const float* __restrict__ cb4,
    const float* __restrict__ cb8, const float* __restrict__ cb16,
    const float* __restrict__ cb32,
    const float* __restrict__ fusion_w, _Float16* __restrict__ xgh) {
  const int lane = threadIdx.x & 31;
  const int tt = blockIdx.x, nt = blockIdx.y, bk = blockIdx.z;
  const int band = bk % KBANDS, b = bk / KBANDS;
  float f0 = fusion_w[band * 3 + 0], f1 = fusion_w[band * 3 + 1], f2 = fusion_w[band * 3 + 2];
  float fm = fmaxf(f0, fmaxf(f1, f2));
  float e0 = __expf(f0 - fm), e1 = __expf(f1 - fm), e2 = __expf(f2 - fm);
  float esum = e0 + e1 + e2;
  const float wgt[3] = {e0 / esum, e1 / esum, e2 / esum};
  const int g = (band <= 5) ? 0 : (band <= 14 ? 1 : 2);
  const int scg[3][3] = {{8, 16, 32}, {4, 8, 16}, {2, 4, 8}};
  const size_t woff[5] = {0, 32768, 98304, 229376, 491520}; // halfs
  const float* cbs[5] = {cb2, cb4, cb8, cb16, cb32};
  const _Float16* inp = h1h + (size_t)bk * NC * TT;
  const int tcol = tt * 16 + (lane & 15);
  const int hi8 = ((lane >> 4) << 3);
  const int rb = nt * 16 + ((lane >> 4) << 3);
  float total[8] = {0, 0, 0, 0, 0, 0, 0, 0};
  for (int q3 = 0; q3 < 3; ++q3) {
    const int s = scg[g][q3];
    const int si = (s == 2) ? 0 : (s == 4) ? 1 : (s == 8) ? 2 : (s == 16) ? 3 : 4;
    const _Float16* W = wc + woff[si] + (size_t)(nt * 16) * (size_t)(NC * s);
    const int off = s >> 1;
    v8f acc = {};
    for (int kk = 0; kk < 4 * s; ++kk) {
      const int j = kk >> 2;
      const int c0 = (kk & 3) << 5;
      v16h a = ld_frag(W + (size_t)j * NC + c0, NC * s, lane);
      const int tpos = tcol + j - off;
      const bool ok = (tpos >= 0) && (tpos < TT);
      v16h bf;
#pragma unroll
      for (int i2 = 0; i2 < 16; ++i2) {
        int kidx = (i2 < 8) ? (hi8 + i2) : (16 + hi8 + i2 - 8);
        bf[i2] = ok ? inp[(size_t)(c0 + kidx) * TT + tpos] : (_Float16)0.f;
      }
      acc = wmma16(a, bf, acc);
    }
    const float* bias = cbs[si];
#pragma unroll
    for (int i = 0; i < 8; ++i) total[i] += wgt[q3] * (acc[i] + bias[rb + i]);
  }
#pragma unroll
  for (int i = 0; i < 8; ++i)
    xgh[(((size_t)b * TT + tcol) * KBANDS + band) * NC + (rb + i)] = (_Float16)total[i];
}

// ================= stage 3: harmonic graph attention =================

__global__ __launch_bounds__(256) void dbt_adj(const float* __restrict__ be,
                                               float* __restrict__ Amask) {
  __shared__ float e[30 * 64];
  __shared__ float sim[900];
  __shared__ float Af[900];
  const int tid = threadIdx.x;
  if (tid < 30) {
    float s = 0.f;
    for (int d = 0; d < 64; ++d) { float v = be[tid * 64 + d]; s += v * v; }
    float inv = 1.f / (sqrtf(s) + 1e-8f);
    for (int d = 0; d < 64; ++d) e[tid * 64 + d] = be[tid * 64 + d] * inv;
  }
  __syncthreads();
  for (int p = tid; p < 900; p += 256) {
    int i = p / 30, j = p % 30;
    float d = 0.f;
    for (int c = 0; c < 64; ++c) d += e[i * 64 + c] * e[j * 64 + c];
    sim[p] = d; Af[p] = 0.f;
  }
  __syncthreads();
  if (tid < 30) {
    float loc[30];
    for (int j = 0; j < 30; ++j) loc[j] = sim[tid * 30 + j];
    for (int it = 0; it < 5; ++it) {
      int bi = 0; float bv = loc[0];
      for (int j = 1; j < 30; ++j) if (loc[j] > bv) { bv = loc[j]; bi = j; }
      Af[tid * 30 + bi] = bv;
      loc[bi] = -1e30f;
    }
  }
  __syncthreads();
  for (int p = tid; p < 900; p += 256) {
    int i = p / 30, j = p % 30;
    Amask[p] = 0.5f * (Af[i * 30 + j] + Af[j * 30 + i]);
  }
}

__global__ __launch_bounds__(32) void dbt_gproj(const _Float16* __restrict__ xgh,
    const _Float16* __restrict__ gwT,
    const float* __restrict__ gbq, const float* __restrict__ gbk,
    const float* __restrict__ gbv,
    _Float16* __restrict__ gq, _Float16* __restrict__ gk, _Float16* __restrict__ gv) {
  const int lane = threadIdx.x & 31;
  const int rt = blockIdx.x, nt = blockIdx.y, which = blockIdx.z;
  const _Float16* A0 = xgh + (size_t)rt * 16 * NC;
  const _Float16* BT = gwT + (size_t)which * NC * NC + (size_t)nt * 16 * NC;
  v8f acc = {};
#pragma unroll
  for (int ks = 0; ks < 4; ++ks)
    acc = wmma16(ld_frag(A0 + ks * 32, NC, lane), ld_frag(BT + ks * 32, NC, lane), acc);
  const float* bias = (which == 0) ? gbq : (which == 1) ? gbk : gbv;
  _Float16* out = (which == 0) ? gq : (which == 1) ? gk : gv;
  const int col = nt * 16 + (lane & 15);
  const int rb = rt * 16 + ((lane >> 4) << 3);
  const float bv = bias[col];
#pragma unroll
  for (int i = 0; i < 8; ++i)
    out[(size_t)(rb + i) * NC + col] = (_Float16)(acc[i] + bv);
}

__global__ __launch_bounds__(128) void dbt_gscore(const _Float16* __restrict__ gq,
    const _Float16* __restrict__ gk, float* __restrict__ gS) {
  const int m = blockIdx.x, h = blockIdx.y, tid = threadIdx.x;
  const float scale = 0.17677669529663687f; // 1/sqrt(32)
  for (int p = tid; p < 900; p += 128) {
    int i = p / 30, j = p % 30;
    const _Float16* qr = gq + ((size_t)m * 30 + i) * NC + h * 32;
    const _Float16* kr = gk + ((size_t)m * 30 + j) * NC + h * 32;
    float d = 0.f;
    for (int c = 0; c < 32; ++c) d += (float)qr[c] * (float)kr[c];
    gS[(((size_t)m * 4 + h) * 30 + i) * 30 + j] = d * scale;
  }
}

__global__ __launch_bounds__(128) void dbt_gattn(const float* __restrict__ gS,
    const _Float16* __restrict__ gv, const float* __restrict__ Amask,
    _Float16* __restrict__ ogh) {
  __shared__ float sc[900];
  __shared__ float msk[900];
  const int m = blockIdx.x, h = blockIdx.y, tid = threadIdx.x;
  const float* Sp = gS + ((size_t)m * 4 + h) * 900;
  for (int p = tid; p < 900; p += 128) { sc[p] = Sp[p]; msk[p] = Amask[p]; }
  __syncthreads();
  if (tid < 30) {
    float mx = -1e30f; bool any = false;
    for (int j = 0; j < 30; ++j)
      if (msk[tid * 30 + j] != 0.f) { any = true; mx = fmaxf(mx, sc[tid * 30 + j]); }
    float tmp[30]; float sum = 0.f;
    for (int j = 0; j < 30; ++j) {
      float e = (msk[tid * 30 + j] != 0.f) ? __expf(sc[tid * 30 + j] - mx) : 0.f;
      tmp[j] = e; sum += e;
    }
    float inv = (any && sum > 0.f) ? (1.f / sum) : 0.f; // nan_to_num -> 0
    for (int j = 0; j < 30; ++j) sc[tid * 30 + j] = tmp[j] * inv;
  }
  __syncthreads();
  for (int p = tid; p < 960; p += 128) {
    int i = p >> 5, d = p & 31;
    float acc = 0.f;
    for (int j = 0; j < 30; ++j)
      acc += sc[i * 30 + j] * (float)gv[((size_t)m * 30 + j) * NC + h * 32 + d];
    ogh[((size_t)m * 30 + i) * NC + h * 32 + d] = (_Float16)acc;
  }
}

__global__ __launch_bounds__(32) void dbt_gout(const _Float16* __restrict__ ogh,
    const _Float16* __restrict__ gwT, const float* __restrict__ gbo,
    float* __restrict__ out) {
  const int lane = threadIdx.x & 31;
  const int rt = blockIdx.x, nt = blockIdx.y;
  const _Float16* A0 = ogh + (size_t)rt * 16 * NC;
  const _Float16* BT = gwT + (size_t)3 * NC * NC + (size_t)nt * 16 * NC;
  v8f acc = {};
#pragma unroll
  for (int ks = 0; ks < 4; ++ks)
    acc = wmma16(ld_frag(A0 + ks * 32, NC, lane), ld_frag(BT + ks * 32, NC, lane), acc);
  const int col = nt * 16 + (lane & 15);
  const int rb = rt * 16 + ((lane >> 4) << 3);
  const float bias = gbo[col];
#pragma unroll
  for (int i = 0; i < 8; ++i) {
    int r = rb + i;
    int m = r / 30, band = r % 30;
    int b = m >> 9, t = m & 511;
    out[(((size_t)b * NC + col) * TT + t) * KBANDS + band] = acc[i] + bias;
  }
}

// ================= host orchestration =================

extern "C" void kernel_launch(void* const* d_in, const int* in_sizes, int n_in,
                              void* d_out, int out_size, void* d_ws, size_t ws_size,
                              hipStream_t stream) {
  (void)in_sizes; (void)n_in; (void)out_size; (void)ws_size;
  const float* x    = (const float*)d_in[0];
  const float* Wq   = (const float*)d_in[1];
  const float* Wk   = (const float*)d_in[2];
  const float* Wv   = (const float*)d_in[3];
  const float* Wo   = (const float*)d_in[4];
  const float* bo   = (const float*)d_in[5];
  const float* lw1  = (const float*)d_in[6];
  const float* lb1  = (const float*)d_in[7];
  const float* lw2  = (const float*)d_in[8];
  const float* lb2  = (const float*)d_in[9];
  const float* cw2  = (const float*)d_in[10];
  const float* cb2  = (const float*)d_in[11];
  const float* cw4  = (const float*)d_in[12];
  const float* cb4  = (const float*)d_in[13];
  const float* cw8  = (const float*)d_in[14];
  const float* cb8  = (const float*)d_in[15];
  const float* cw16 = (const float*)d_in[16];
  const float* cb16 = (const float*)d_in[17];
  const float* cw32 = (const float*)d_in[18];
  const float* cb32 = (const float*)d_in[19];
  const float* fusion_w   = (const float*)d_in[20];
  const float* band_embed = (const float*)d_in[21];
  const float* gWq = (const float*)d_in[22];
  const float* gbq = (const float*)d_in[23];
  const float* gWk = (const float*)d_in[24];
  const float* gbk = (const float*)d_in[25];
  const float* gWv = (const float*)d_in[26];
  const float* gbv = (const float*)d_in[27];
  const float* gWo = (const float*)d_in[28];
  const float* gbo = (const float*)d_in[29];
  float* out = (float*)d_out;

  char* ws = (char*)d_ws;
  _Float16* wqT = (_Float16*)(ws + o_wqT);
  _Float16* wkT = (_Float16*)(ws + o_wkT);
  _Float16* wvT = (_Float16*)(ws + o_wvT);
  _Float16* woT = (_Float16*)(ws + o_woT);
  _Float16* gwT = (_Float16*)(ws + o_gwT);
  _Float16* wc  = (_Float16*)(ws + o_wc);
  float* Amask  = (float*)(ws + o_amask);
  float* lam    = (float*)(ws + o_lam);
  float* rstat  = (float*)(ws + o_rstat);
  _Float16* xh  = (_Float16*)(ws + o_xh);
  _Float16* qh  = (_Float16*)(ws + o_qh);
  _Float16* kh  = (_Float16*)(ws + o_kh);
  _Float16* vT  = (_Float16*)(ws + o_vT);
  float* S1     = (float*)(ws + o_S1);
  float* S2     = (float*)(ws + o_S2);
  _Float16* P1  = (_Float16*)(ws + o_P1);
  _Float16* P2  = (_Float16*)(ws + o_P2);
  _Float16* obh = (_Float16*)(ws + o_obh);
  _Float16* h1h = (_Float16*)(ws + o_h1h);
  _Float16* xgh = (_Float16*)(ws + o_xgh);
  _Float16* gq  = (_Float16*)(ws + o_gq);
  _Float16* gk  = (_Float16*)(ws + o_gk);
  _Float16* gv  = (_Float16*)(ws + o_gv);
  float* gS     = (float*)(ws + o_gS);
  _Float16* ogh = (_Float16*)(ws + o_ogh);

  auto nb = [](int total) { return (unsigned)((total + 255) / 256); };

  // weight packing (transpose + f32->f16)
  dbt_tcvt<<<nb(30 * 128 * 128), 256, 0, stream>>>(Wq, wqT, 30, 128, 128);
  dbt_tcvt<<<nb(30 * 128 * 128), 256, 0, stream>>>(Wk, wkT, 30, 128, 128);
  dbt_tcvt<<<nb(30 * 128 * 64), 256, 0, stream>>>(Wv, wvT, 30, 128, 64);
  dbt_tcvt<<<nb(30 * 64 * 128), 256, 0, stream>>>(Wo, woT, 30, 64, 128);
  dbt_tcvt<<<nb(128 * 128), 256, 0, stream>>>(gWq, gwT + 0 * 16384, 1, 128, 128);
  dbt_tcvt<<<nb(128 * 128), 256, 0, stream>>>(gWk, gwT + 1 * 16384, 1, 128, 128);
  dbt_tcvt<<<nb(128 * 128), 256, 0, stream>>>(gWv, gwT + 2 * 16384, 1, 128, 128);
  dbt_tcvt<<<nb(128 * 128), 256, 0, stream>>>(gWo, gwT + 3 * 16384, 1, 128, 128);
  dbt_tcvt<<<nb(128 * 128 * 2), 256, 0, stream>>>(cw2, wc + 0, 128, 128, 2);
  dbt_tcvt<<<nb(128 * 128 * 4), 256, 0, stream>>>(cw4, wc + 32768, 128, 128, 4);
  dbt_tcvt<<<nb(128 * 128 * 8), 256, 0, stream>>>(cw8, wc + 98304, 128, 128, 8);
  dbt_tcvt<<<nb(128 * 128 * 16), 256, 0, stream>>>(cw16, wc + 229376, 128, 128, 16);
  dbt_tcvt<<<nb(128 * 128 * 32), 256, 0, stream>>>(cw32, wc + 491520, 128, 128, 32);

  dbt_pack_x<<<nb(KB * TT * NC), 256, 0, stream>>>(x, xh);
  dbt_adj<<<1, 256, 0, stream>>>(band_embed, Amask);

  // stage 1: batched QKV, then per-(band,batch) attention reusing score buffers
  dbt_qkv<<<dim3(32, 20, 120), 32, 0, stream>>>(xh, wqT, wkT, wvT, qh, kh, vT);
  for (int kb = 0; kb < KB; ++kb) {
    dbt_scores<<<dim3(32, 32, 2), 32, 0, stream>>>(qh, kh, S1, S2, kb);
    dbt_softmax<<<dim3(512, 2), 256, 0, stream>>>(S1, S2, P1, P2, rstat);
    dbt_lambda<<<1, 256, 0, stream>>>(rstat, lw1, lb1, lw2, lb2, lam, kb);
    dbt_pv<<<dim3(32, 4), 32, 0, stream>>>(P1, P2, vT, lam, obh, kb);
    dbt_wo<<<dim3(32, 8), 32, 0, stream>>>(obh, woT, bo, h1h, kb);
  }

  // stage 2: fused multiscale conv (implicit GEMM) -> xgh
  dbt_conv<<<dim3(32, 8, 120), 32, 0, stream>>>(h1h, wc, cb2, cb4, cb8, cb16, cb32,
                                                fusion_w, xgh);

  // stage 3: graph attention across bands
  dbt_gproj<<<dim3(RG / 16, 8, 3), 32, 0, stream>>>(xgh, gwT, gbq, gbk, gbv, gq, gk, gv);
  dbt_gscore<<<dim3(MTOK, 4), 128, 0, stream>>>(gq, gk, gS);
  dbt_gattn<<<dim3(MTOK, 4), 128, 0, stream>>>(gS, gv, Amask, ogh);
  dbt_gout<<<dim3(RG / 16, 8), 32, 0, stream>>>(ogh, gwT, gbo, out);
}
